// LSTMCell_81518479278547
// MI455X (gfx1250) — matrix operand order repaired and checked by reading, not compiled, not run on hardware
//
#include <hip/hip_runtime.h>
#include <hip/hip_bf16.h>

typedef __attribute__((ext_vector_type(16))) __bf16   v16bf;
typedef __attribute__((ext_vector_type(2)))  __bf16   v2bf;
typedef __attribute__((ext_vector_type(8)))  float    v8f;
typedef __attribute__((ext_vector_type(4)))  unsigned uint4v;

#define HDIM   2048
#define BATCH  4096
#define KT     32
#define TM     64
#define TN     64
#define NSTEPS 128   // (2048/32) x-phase + (2048/32) h-phase

struct Params {
    const float* x;  const float* hs; const float* cs;
    const float* wx[4]; const float* wh[4];
    const float* bx[4]; const float* bh[4];
    float* hout; float* cout;
};

// fp32 pair -> packed bf16 (hardware v_cvt_pk_bf16_f32)
__device__ __forceinline__ v2bf cvt2(float lo, float hi) {
#if __has_builtin(__builtin_amdgcn_cvt_pk_bf16_f32)
    return __builtin_amdgcn_cvt_pk_bf16_f32(lo, hi);
#else
    v2bf r; r[0] = (__bf16)lo; r[1] = (__bf16)hi; return r;
#endif
}

union Pack8 { v2bf p[4]; uint4v q; };

// split workgroup barrier: producer side only needs DS stores flushed;
// outstanding global loads (to registers) must NOT be drained here.
__device__ __forceinline__ void wg_signal() {
    asm volatile("s_wait_dscnt 0x0\n\ts_barrier_signal -1" ::: "memory");
}
__device__ __forceinline__ void wg_wait() {
    asm volatile("s_barrier_wait -1" ::: "memory");
}

__device__ __forceinline__ float fsigmoid(float x) {
    return __builtin_amdgcn_rcpf(1.0f + __expf(-x));
}
// overflow-safe tanh: sign(x) * (1-e)/(1+e), e = exp(-2|x|) in (0,1]
__device__ __forceinline__ float ftanh(float x) {
    float e = __expf(-2.0f * __builtin_fabsf(x));
    float t = (1.0f - e) * __builtin_amdgcn_rcpf(1.0f + e);
    return __builtin_copysignf(t, x);
}

__global__ __launch_bounds__(256, 2) void lstm_wmma_bf16(Params p) {
    __shared__ __align__(16) unsigned short sA[2][TM * KT];        //  8 KB
    __shared__ __align__(16) unsigned short sB[2][4][TN * KT];     // 32 KB

    const int tid   = threadIdx.x;
    const int lane  = tid & 31;
    const int wid   = tid >> 5;
    const int waveM = wid >> 2;          // 0..1  -> 32 rows each
    const int waveN = wid & 3;           // 0..3  -> 16 cols each
    const int m0    = blockIdx.y * TM;
    const int n0    = blockIdx.x * TN;

    // cooperative staging: thread -> (row 0..63, 8-float chunk 0..3)
    const int lr = tid >> 2;
    const int lc = (tid & 3) * 8;

    v8f acc[2][4];
#pragma unroll
    for (int mt = 0; mt < 2; ++mt)
#pragma unroll
        for (int g = 0; g < 4; ++g) acc[mt][g] = 0.0f;

    float4 av[2];
    float4 wv[4][2];

    auto fetch = [&](int step) {
        const int ph = step >> 6;                 // 0: x-phase, 1: h-phase
        const int kk = (step & 63) * KT;
        const float* A = ph ? p.hs : p.x;
        const float4* ap =
            reinterpret_cast<const float4*>(A + (size_t)(m0 + lr) * HDIM + kk + lc);
        av[0] = ap[0]; av[1] = ap[1];
#pragma unroll
        for (int g = 0; g < 4; ++g) {
            const float* W = ph ? p.wh[g] : p.wx[g];
            const float4* wp =
                reinterpret_cast<const float4*>(W + (size_t)(n0 + lr) * HDIM + kk + lc);
            wv[g][0] = wp[0]; wv[g][1] = wp[1];
        }
    };

    auto stage = [&](int buf) {
        Pack8 pa;
        pa.p[0] = cvt2(av[0].x, av[0].y); pa.p[1] = cvt2(av[0].z, av[0].w);
        pa.p[2] = cvt2(av[1].x, av[1].y); pa.p[3] = cvt2(av[1].z, av[1].w);
        *reinterpret_cast<uint4v*>(&sA[buf][lr * KT + lc]) = pa.q;
#pragma unroll
        for (int g = 0; g < 4; ++g) {
            Pack8 pw;
            pw.p[0] = cvt2(wv[g][0].x, wv[g][0].y); pw.p[1] = cvt2(wv[g][0].z, wv[g][0].w);
            pw.p[2] = cvt2(wv[g][1].x, wv[g][1].y); pw.p[3] = cvt2(wv[g][1].z, wv[g][1].w);
            *reinterpret_cast<uint4v*>(&sB[buf][g][lr * KT + lc]) = pw.q;
        }
    };

    union Frag { uint4v q[2]; v16bf v; };

    // per-lane fragment coordinates (wave32; contiguous K order, k = 16*(lane/16) + s)
    const int mrow  = waveM * 32 + (lane & 15);   // A: row within tile
    const int abase = (lane >> 4) * 16;           // A: contiguous K base
    const int nrow  = waveN * 16 + (lane & 15);   // B: W-row (output col)
    const int kb    = (lane >> 4) * 16;           // B: contiguous K base

    auto compute = [&](int buf) {
        Frag bfr[4];
#pragma unroll
        for (int g = 0; g < 4; ++g) {
            const uint4v* bp =
                reinterpret_cast<const uint4v*>(&sB[buf][g][nrow * KT + kb]);
            bfr[g].q[0] = bp[0]; bfr[g].q[1] = bp[1];
        }
#pragma unroll
        for (int mt = 0; mt < 2; ++mt) {
            Frag af;
            const unsigned short* ap = &sA[buf][(mrow + mt * 16) * KT + abase];
            af.q[0] = *reinterpret_cast<const uint4v*>(ap);        // K: b..b+7
            af.q[1] = *reinterpret_cast<const uint4v*>(ap + 8);    // K: b+8..b+15
#pragma unroll
            for (int g = 0; g < 4; ++g) {
                acc[mt][g] = __builtin_amdgcn_wmma_f32_16x16x32_bf16(
                    false, af.v, false, bfr[g].v, (short)0, acc[mt][g], false, false);
            }
        }
    };

    // software-pipelined main loop, split barrier:
    //   signal after staging (DS flushed), wait before consuming;
    //   global prefetch for step+1 stays in flight across the barrier.
    fetch(0);
    stage(0);
    wg_signal();
    int cur = 0;
    for (int step = 0; step < NSTEPS; ++step) {
        if (step + 1 < NSTEPS) fetch(step + 1);
        wg_wait();
        compute(cur);
        if (step + 1 < NSTEPS) { stage(cur ^ 1); wg_signal(); }
        cur ^= 1;
    }

    // epilogue: bias + LSTM gate fusion, write h', c'
    const int nG = n0 + waveN * 16 + (lane & 15);
    float bias[4];
#pragma unroll
    for (int g = 0; g < 4; ++g) bias[g] = p.bx[g][nG] + p.bh[g][nG];

    const int mbase = m0 + waveM * 32 + 8 * (lane >> 4);
#pragma unroll
    for (int mt = 0; mt < 2; ++mt) {
#pragma unroll
        for (int r = 0; r < 8; ++r) {
            const int m = mbase + mt * 16 + r;
            const float ig = fsigmoid(acc[mt][0][r] + bias[0]);
            const float fg = fsigmoid(acc[mt][1][r] + bias[1]);
            const float gg = ftanh  (acc[mt][2][r] + bias[2]);
            const float og = fsigmoid(acc[mt][3][r] + bias[3]);
            const size_t idx = (size_t)m * HDIM + nG;
            const float cnew = fg * p.cs[idx] + ig * gg;
            p.cout[idx] = cnew;
            p.hout[idx] = og * ftanh(cnew);
        }
    }
}

extern "C" void kernel_launch(void* const* d_in, const int* in_sizes, int n_in,
                              void* d_out, int out_size, void* d_ws, size_t ws_size,
                              hipStream_t stream) {
    (void)in_sizes; (void)n_in; (void)d_ws; (void)ws_size; (void)out_size;
    Params prm;
    prm.x  = (const float*)d_in[0];
    prm.hs = (const float*)d_in[1];
    prm.cs = (const float*)d_in[2];
    for (int g = 0; g < 4; ++g) {
        prm.wx[g] = (const float*)d_in[3 + 4 * g];
        prm.bx[g] = (const float*)d_in[4 + 4 * g];
        prm.wh[g] = (const float*)d_in[5 + 4 * g];
        prm.bh[g] = (const float*)d_in[6 + 4 * g];
    }
    prm.hout = (float*)d_out;
    prm.cout = (float*)d_out + (size_t)BATCH * HDIM;

    dim3 grid(HDIM / TN, BATCH / TM);   // (32, 64)
    dim3 block(256);
    lstm_wmma_bf16<<<grid, block, 0, stream>>>(prm);
}
